// Alpha_DC_70987219469023
// MI455X (gfx1250) — compile-verified, matrix operand order checked by reading, not run on hardware
//
#include <hip/hip_runtime.h>
#include <hip/hip_bf16.h>

// ---------------------------------------------------------------------------
// Alpha-DCov (dim=1024, M=512, B=32), MI455X / gfx1250 wave32 WMMA pipeline.
//
//  0) split     : X (f32) -> Xhi/Xlo bf16 (hi/lo split, one-time conversion)
//  1) gram_gemm : S[b] = X[b] @ X[b]^T * 1/(2M) via 3x v_wmma_f32_16x16x32_bf16
//                 per K=32 step (hi*hi + hi*lo + lo*hi ~= f32 accuracy).
//                 Double-buffered LDS fed by GLOBAL_LOAD_ASYNC_TO_LDS_B128
//                 (ASYNCcnt) so DMA overlaps WMMA with zero staging registers.
//  2) diag      : d[b,i] = S[b,i,i]
//  3) row_pass  : rowmean[b,i] = mean_j (max(d_i+d_j-2 s_ij,0)+eps)^0.4
//  4) tot_pass  : totmean[b]   = mean_i rowmean[b,i]
//  5) finalize  : out = f(s_ij) - rm_i - rm_j + tm, packed upper-tri rows
//
// Workspace: S (128MB f32) + Xhi/Xlo (32MB bf16 each) + diag/rowmean/totmean.
// ---------------------------------------------------------------------------

#define DIMN 1024
#define MK   512
#define NBATCH 32
#define TRI  (DIMN * (DIMN + 1) / 2)   // 524800
#define EPSV 1e-5f
#define ALPHAV 0.4f

typedef __attribute__((ext_vector_type(16))) __bf16 bf16x16;
typedef __attribute__((ext_vector_type(8)))  __bf16 bf16x8;
typedef __attribute__((ext_vector_type(8)))  float  f32x8;
typedef __attribute__((ext_vector_type(4)))  float  f32x4;
typedef __attribute__((ext_vector_type(4)))  int    i32x4;

constexpr int BM = 128;     // C tile rows per workgroup
constexpr int BN = 128;     // C tile cols per workgroup
constexpr int BK = 32;      // K step (bf16 WMMA depth)
constexpr int LSTRH = 40;   // LDS row stride in bf16 units (80B: aligned + conflict-free)

#if defined(__has_builtin)
#  if __has_builtin(__builtin_amdgcn_global_load_async_to_lds_b128)
#    define USE_ASYNC 1
#  else
#    define USE_ASYNC 0
#  endif
#else
#  define USE_ASYNC 0
#endif

// 16B global -> LDS copy; async (ASYNCcnt-tracked) when available.
// Builtin signature (from hipcc diagnostic): param0 is `int4 __device__*`
// (global AS, non-const). Cast through (void*) to drop const, then into the
// exact address-space-qualified int4 pointer types.
__device__ __forceinline__ void copy16_g2l(const uint4* __restrict__ g, __bf16* l) {
#if USE_ASYNC
  __builtin_amdgcn_global_load_async_to_lds_b128(
      (__attribute__((address_space(1))) i32x4*)(void*)g,
      (__attribute__((address_space(3))) i32x4*)(void*)l, 0, 0);
#else
  *(uint4*)l = *g;
#endif
}

__device__ __forceinline__ void wait_async() {
#if USE_ASYNC
#  if __has_builtin(__builtin_amdgcn_s_wait_asynccnt)
  __builtin_amdgcn_s_wait_asynccnt(0);
#  else
  asm volatile("s_wait_asynccnt 0x0" ::: "memory");
#  endif
#endif
}

// ---------------------------------------------------------------------------
// 0) hi/lo bf16 split (one-time; removes all conversion VALU from GEMM loop)
// ---------------------------------------------------------------------------
__global__ __launch_bounds__(256) void split_kernel(const float* __restrict__ X,
                                                    __bf16* __restrict__ Xhi,
                                                    __bf16* __restrict__ Xlo) {
  size_t g = ((size_t)blockIdx.x * 256 + threadIdx.x) * 8;
  f32x4 v0 = *(const f32x4*)(X + g);
  f32x4 v1 = *(const f32x4*)(X + g + 4);
  bf16x8 h, l;
#pragma unroll
  for (int i = 0; i < 4; ++i) {
    __bf16 hh = (__bf16)v0[i];
    h[i] = hh;
    l[i] = (__bf16)(v0[i] - (float)hh);
  }
#pragma unroll
  for (int i = 0; i < 4; ++i) {
    __bf16 hh = (__bf16)v1[i];
    h[4 + i] = hh;
    l[4 + i] = (__bf16)(v1[i] - (float)hh);
  }
  *(bf16x8*)(Xhi + g) = h;
  *(bf16x8*)(Xlo + g) = l;
}

// ---------------------------------------------------------------------------
// 1) Gram GEMM
// ---------------------------------------------------------------------------

// Issue the 8 x 16B copies this thread owns for one k-slab (128x32 bf16 x4 tiles).
__device__ __forceinline__ void stage_issue(const uint4* __restrict__ Ghi,
                                            const uint4* __restrict__ Glo,
                                            __bf16* Ah, __bf16* Al,
                                            __bf16* Bh, __bf16* Bl,
                                            int rowA, int rowB, int tid, int ksq) {
#pragma unroll
  for (int u = 0; u < 2; ++u) {
    int c   = tid + u * 256;
    int row = c >> 2;        // 0..127
    int q   = c & 3;         // 16B chunk within 64B k-slab
    size_t ia = (size_t)(rowA + row) * (MK / 8) + ksq + q;
    size_t ib = (size_t)(rowB + row) * (MK / 8) + ksq + q;
    int lo = row * LSTRH + q * 8;
    copy16_g2l(Ghi + ia, &Ah[lo]);
    copy16_g2l(Glo + ia, &Al[lo]);
    copy16_g2l(Ghi + ib, &Bh[lo]);
    copy16_g2l(Glo + ib, &Bl[lo]);
  }
}

// Load one 16x32 bf16 fragment from an LDS bf16 tile.
// A-matrix layout (ISA): lanes 0-15 hold K 0..7 & 16..23; lanes 16-31 hold
// K 8..15 & 24..31. For a Gram matrix the B fragment (K x 16 cols = rows of X)
// has the identical per-lane pattern, so one loader serves both operands.
__device__ __forceinline__ bf16x16 load_frag(const __bf16* tile, int r0, int lane) {
  const int m    = lane & 15;
  const int koff = (lane >> 4) << 3;  // 0 or 8
  const __bf16* p = tile + (r0 + m) * LSTRH + koff;
  union { bf16x16 v; uint4 u[2]; } f;
  f.u[0] = *(const uint4*)p;          // K koff..koff+7
  f.u[1] = *(const uint4*)(p + 16);   // K koff+16..koff+23
  return f.v;
}

__global__ __launch_bounds__(256, 2) void gram_gemm_kernel(const __bf16* __restrict__ Xhi,
                                                           const __bf16* __restrict__ Xlo,
                                                           float* __restrict__ S) {
  // Double-buffered bf16 tiles: 2 x 4 x 10,240B = 80KB LDS.
  __shared__ __bf16 AhS[2][BM * LSTRH];
  __shared__ __bf16 AlS[2][BM * LSTRH];
  __shared__ __bf16 BhS[2][BN * LSTRH];
  __shared__ __bf16 BlS[2][BN * LSTRH];

  const int b     = blockIdx.y;
  const int tileM = (blockIdx.x >> 3) * BM;
  const int tileN = (blockIdx.x & 7) * BN;
  const uint4* __restrict__ Ghi = (const uint4*)(Xhi + (size_t)b * DIMN * MK);
  const uint4* __restrict__ Glo = (const uint4*)(Xlo + (size_t)b * DIMN * MK);
  float* __restrict__ Sb = S + (size_t)b * DIMN * DIMN;

  const int tid  = threadIdx.x;
  const int wave = tid >> 5;           // 0..7
  const int lane = tid & 31;
  const int wRow = (wave >> 1) * 32;   // 0,32,64,96 within tile
  const int wCol = (wave & 1) * 64;    // 0,64 within tile

  f32x8 acc[2][4] = {};

  // Prologue: fill buffer 0.
  stage_issue(Ghi, Glo, AhS[0], AlS[0], BhS[0], BlS[0], tileM, tileN, tid, 0);
  wait_async();
  __syncthreads();

  for (int ks = 0; ks < MK; ks += BK) {
    const int cur = (ks >> 5) & 1;
    const int nxt = cur ^ 1;

    // Kick off DMA for the next k-slab; it flows while WMMAs run on `cur`.
    if (ks + BK < MK)
      stage_issue(Ghi, Glo, AhS[nxt], AlS[nxt], BhS[nxt], BlS[nxt],
                  tileM, tileN, tid, (ks + BK) >> 3);

    bf16x16 ah[2], al[2];
#pragma unroll
    for (int i = 0; i < 2; ++i) {
      ah[i] = load_frag(AhS[cur], wRow + i * 16, lane);
      al[i] = load_frag(AlS[cur], wRow + i * 16, lane);
    }
#pragma unroll
    for (int j = 0; j < 4; ++j) {
      bf16x16 bh = load_frag(BhS[cur], wCol + j * 16, lane);
      bf16x16 bl = load_frag(BlS[cur], wCol + j * 16, lane);
#pragma unroll
      for (int i = 0; i < 2; ++i) {
        // hi*hi + hi*lo + lo*hi  (f32-split bf16 emulation; lo*lo negligible)
        acc[i][j] = __builtin_amdgcn_wmma_f32_16x16x32_bf16(
            false, ah[i], false, bh, (short)0, acc[i][j], false, false);
        acc[i][j] = __builtin_amdgcn_wmma_f32_16x16x32_bf16(
            false, ah[i], false, bl, (short)0, acc[i][j], false, false);
        acc[i][j] = __builtin_amdgcn_wmma_f32_16x16x32_bf16(
            false, al[i], false, bh, (short)0, acc[i][j], false, false);
      }
    }

    wait_async();      // this wave's copies into `nxt` have landed in LDS
    __syncthreads();   // all waves: done reading `cur`, done writing `nxt`
  }

  // Store: C/D layout -> VGPR e: lanes 0-15 M=e, lanes 16-31 M=e+8; N = lane&15.
  const float scale = 1.0f / (2.0f * (float)MK);
  const int colL = lane & 15;
  const int rAdd = (lane >> 4) << 3;
#pragma unroll
  for (int i = 0; i < 2; ++i) {
#pragma unroll
    for (int j = 0; j < 4; ++j) {
#pragma unroll
      for (int e = 0; e < 8; ++e) {
        int row = tileM + wRow + i * 16 + rAdd + e;
        int col = tileN + wCol + j * 16 + colL;
        Sb[(size_t)row * DIMN + col] = acc[i][j][e] * scale;
      }
    }
  }
}

// ---------------------------------------------------------------------------
// 2-5) elementwise + reductions (memory-bound, deterministic, atomic-free)
// ---------------------------------------------------------------------------
__global__ __launch_bounds__(256) void diag_kernel(const float* __restrict__ S,
                                                   float* __restrict__ diag) {
  int g = blockIdx.x * 256 + threadIdx.x;   // [0, NBATCH*DIMN)
  int b = g >> 10;
  int i = g & (DIMN - 1);
  diag[g] = S[((size_t)b * DIMN + i) * DIMN + i];
}

__device__ __forceinline__ float dcov_f(float di, float dj, float sij) {
  float t = di + dj - 2.0f * sij;
  t = fmaxf(t, 0.0f) + EPSV;
  return exp2f(ALPHAV * log2f(t));
}

__global__ __launch_bounds__(256) void row_kernel(const float* __restrict__ S,
                                                  const float* __restrict__ diag,
                                                  float* __restrict__ rowmean) {
  const int i = blockIdx.x, b = blockIdx.y;
  const float* __restrict__ Srow = S + ((size_t)b * DIMN + i) * DIMN;
  const float* __restrict__ db   = diag + b * DIMN;
  const float di = db[i];
  float sum = 0.0f;
  for (int j = threadIdx.x; j < DIMN; j += 256)
    sum += dcov_f(di, db[j], Srow[j]);
  __shared__ float red[256];
  red[threadIdx.x] = sum;
  __syncthreads();
  for (int s = 128; s > 0; s >>= 1) {
    if (threadIdx.x < s) red[threadIdx.x] += red[threadIdx.x + s];
    __syncthreads();
  }
  if (threadIdx.x == 0) rowmean[b * DIMN + i] = red[0] * (1.0f / DIMN);
}

__global__ __launch_bounds__(256) void tot_kernel(const float* __restrict__ rowmean,
                                                  float* __restrict__ totmean) {
  const int b = blockIdx.x;
  float s = 0.0f;
  for (int i = threadIdx.x; i < DIMN; i += 256) s += rowmean[b * DIMN + i];
  __shared__ float red[256];
  red[threadIdx.x] = s;
  __syncthreads();
  for (int k = 128; k > 0; k >>= 1) {
    if (threadIdx.x < k) red[threadIdx.x] += red[threadIdx.x + k];
    __syncthreads();
  }
  if (threadIdx.x == 0) totmean[b] = red[0] * (1.0f / DIMN);
}

__global__ __launch_bounds__(256) void tri_kernel(const float* __restrict__ S,
                                                  const float* __restrict__ diag,
                                                  const float* __restrict__ rowmean,
                                                  const float* __restrict__ totmean,
                                                  float* __restrict__ out) {
  const int i = blockIdx.x, b = blockIdx.y;
  const float* __restrict__ Srow = S + ((size_t)b * DIMN + i) * DIMN;
  const float* __restrict__ db   = diag + b * DIMN;
  const float* __restrict__ rm   = rowmean + b * DIMN;
  const float di = db[i];
  const float ri = rm[i];
  const float tm = totmean[b];
  // packed row base: i*DIMN - i*(i-1)/2 ; entry (i,j) at base + (j-i)
  const int base = i * DIMN - ((i * (i - 1)) >> 1);
  float* __restrict__ ob = out + (size_t)b * TRI + (base - i);
  for (int j = i + threadIdx.x; j < DIMN; j += 256) {
    float f = dcov_f(di, db[j], Srow[j]);
    ob[j] = f - ri - rm[j] + tm;
  }
}

extern "C" void kernel_launch(void* const* d_in, const int* in_sizes, int n_in,
                              void* d_out, int out_size, void* d_ws, size_t ws_size,
                              hipStream_t stream) {
  (void)in_sizes; (void)n_in; (void)out_size; (void)ws_size;
  const float* X = (const float*)d_in[0];
  float* out = (float*)d_out;

  // Workspace layout: S (f32) | Xhi (bf16) | Xlo (bf16) | diag | rowmean | totmean
  float*  S   = (float*)d_ws;
  __bf16* Xhi = (__bf16*)(S + (size_t)NBATCH * DIMN * DIMN);
  __bf16* Xlo = Xhi + (size_t)NBATCH * DIMN * MK;
  float*  diag    = (float*)(Xlo + (size_t)NBATCH * DIMN * MK);
  float*  rowmean = diag + (size_t)NBATCH * DIMN;
  float*  totmean = rowmean + (size_t)NBATCH * DIMN;

  const int totalX = NBATCH * DIMN * MK;           // 16,777,216 elems
  split_kernel<<<totalX / (256 * 8), 256, 0, stream>>>(X, Xhi, Xlo);

  dim3 gGemm((DIMN / BM) * (DIMN / BN), NBATCH);   // 64 x 32
  gram_gemm_kernel<<<gGemm, 256, 0, stream>>>(Xhi, Xlo, S);

  diag_kernel<<<(NBATCH * DIMN) / 256, 256, 0, stream>>>(S, diag);

  dim3 gRows(DIMN, NBATCH);                        // 1024 x 32
  row_kernel<<<gRows, 256, 0, stream>>>(S, diag, rowmean);

  tot_kernel<<<NBATCH, 256, 0, stream>>>(rowmean, totmean);

  tri_kernel<<<gRows, 256, 0, stream>>>(S, diag, rowmean, totmean, out);
}
